// SleepMultiChannelNet_54752243090050
// MI455X (gfx1250) — matrix operand order repaired and checked by reading, not compiled, not run on hardware
//
#include <hip/hip_runtime.h>
#include <stdint.h>
#include <stddef.h>

// ---------------------------------------------------------------------------
// Types for CDNA5 WMMA (wave32): 16x16x32 bf16 -> f32
// ---------------------------------------------------------------------------
typedef __bf16 bf16_t;
typedef __attribute__((ext_vector_type(16))) bf16_t          v16bf;
typedef __attribute__((ext_vector_type(8)))  float           v8f;
typedef __attribute__((ext_vector_type(16))) unsigned short  v16us;
typedef __attribute__((ext_vector_type(8)))  unsigned short  v8us;

__device__ __forceinline__ unsigned short f2bf_u(float f) {
  union { float f; unsigned int u; } cv; cv.f = f;
  unsigned int u = cv.u;
  u += 0x7fffu + ((u >> 16) & 1u);          // round-to-nearest-even
  return (unsigned short)(u >> 16);
}
__device__ __forceinline__ float bfu2f(unsigned short h) {
  union { unsigned int u; float f; } cv; cv.u = ((unsigned int)h) << 16;
  return cv.f;
}
__device__ __forceinline__ float sigm(float x) { return 1.0f / (1.0f + __expf(-x)); }

// A fragment: 16x32 (MxK) bf16, row-major source with row stride ld (elements).
// Lanes 0-15: M=lane, K = 0..15 ; lanes 16-31: M=lane-16, K = 16..31
__device__ __forceinline__ v16bf load_frag_a(const unsigned short* base, int ld, int lane) {
  int row = lane & 15, half = lane >> 4;
  const unsigned short* p = base + row * ld + half * 16;
  v16us t = *(const v16us*)p;         // K: half*16 .. half*16+15
  return __builtin_bit_cast(v16bf, t);
}

// B fragment: 32x16 (KxN) where B[k][n] = W[n][k], W is [N,K] row-major (stride ld).
// Lanes 0-15: N=lane, K=0..15 ; lanes 16-31: N=lane-16, K=16..31
__device__ __forceinline__ v16bf load_frag_b(const unsigned short* wbase, int ld, int lane) {
  int col = lane & 15, half = lane >> 4;
  const unsigned short* p = wbase + col * ld + half * 16;
  v16us t = *(const v16us*)p;         // 32B contiguous along K
  return __builtin_bit_cast(v16bf, t);
}

__device__ __forceinline__ v8f wmma_bf16(v16bf a, v16bf b, v8f c) {
  return __builtin_amdgcn_wmma_f32_16x16x32_bf16(false, a, false, b, (short)0, c, false, false);
}

// ---------------------------------------------------------------------------
// Generic GEMM: out[M,N] = A[M,K](bf16) @ W[N,K]^T (bf16) + bias
// block = 128 threads (4 waves), each wave owns one 16x16 output tile.
// grid.x = N/16, grid.y = M/64
// ---------------------------------------------------------------------------
__global__ __launch_bounds__(128) void gemm_bf16_kernel(
    const unsigned short* __restrict__ A, int lda,
    const unsigned short* __restrict__ W, int ldw,
    const float* __restrict__ bias,
    unsigned short* __restrict__ outBf, int ldob,
    float* __restrict__ outF, int ldof, int K) {
  int lane = threadIdx.x & 31;
  int wid  = threadIdx.x >> 5;
  int n0 = blockIdx.x * 16;
  int m0 = (blockIdx.y * 4 + wid) * 16;
  v8f acc = {0.f, 0.f, 0.f, 0.f, 0.f, 0.f, 0.f, 0.f};
  for (int k0 = 0; k0 < K; k0 += 32) {
    v16bf a = load_frag_a(A + (size_t)m0 * lda + k0, lda, lane);
    v16bf b = load_frag_b(W + (size_t)n0 * ldw + k0, ldw, lane);
    acc = wmma_bf16(a, b, acc);
  }
  int n = lane & 15, mb = (lane >> 4) * 8;
  int col = n0 + n;
  float bv = bias ? bias[col] : 0.0f;
#pragma unroll
  for (int r = 0; r < 8; ++r) {
    int row = m0 + mb + r;
    float v = acc[r] + bv;
    if (outBf) outBf[(size_t)row * ldob + col] = f2bf_u(v);
    if (outF)  outF[(size_t)row * ldof + col] = v;
  }
}

// ---------------------------------------------------------------------------
// Fused LSTM cell step (B=256, H=256):
//   g = x @ Wih^T + h @ Whh^T + bsum ; gates i,f,g,o ; c,h update.
// Each wave owns a 16x16 (batch x hidden) tile and 4 WMMA accumulator chains.
// grid = (16, 4), block = 128.
// ---------------------------------------------------------------------------
__global__ __launch_bounds__(128) void lstm_cell_kernel(
    const unsigned short* __restrict__ x, int ldx, int Dx,
    const unsigned short* __restrict__ Wih,           // [1024, Dx] bf16
    const unsigned short* __restrict__ hprev, int ldh,// [256, 256] bf16
    const unsigned short* __restrict__ Whh,           // [1024, 256] bf16
    const float* __restrict__ bsum,                   // [1024] = bih + bhh
    const float* __restrict__ cprev,                  // [256,256] f32
    float* __restrict__ cout,                         // [256,256] f32
    unsigned short* __restrict__ hout_bf, int ldhob,  // bf16 h out
    float* __restrict__ hout_f, int ldhof) {          // optional f32 h out
  int lane = threadIdx.x & 31;
  int wid  = threadIdx.x >> 5;
  int n0 = blockIdx.x * 16;
  int m0 = (blockIdx.y * 4 + wid) * 16;
  v8f acc[4];
#pragma unroll
  for (int g = 0; g < 4; ++g) acc[g] = (v8f){0.f,0.f,0.f,0.f,0.f,0.f,0.f,0.f};

  for (int k0 = 0; k0 < Dx; k0 += 32) {
    v16bf a = load_frag_a(x + (size_t)m0 * ldx + k0, ldx, lane);
#pragma unroll
    for (int g = 0; g < 4; ++g) {
      v16bf b = load_frag_b(Wih + (size_t)(g * 256 + n0) * Dx + k0, Dx, lane);
      acc[g] = wmma_bf16(a, b, acc[g]);
    }
  }
  for (int k0 = 0; k0 < 256; k0 += 32) {
    v16bf a = load_frag_a(hprev + (size_t)m0 * ldh + k0, ldh, lane);
#pragma unroll
    for (int g = 0; g < 4; ++g) {
      v16bf b = load_frag_b(Whh + (size_t)(g * 256 + n0) * 256 + k0, 256, lane);
      acc[g] = wmma_bf16(a, b, acc[g]);
    }
  }

  int n = lane & 15, mb = (lane >> 4) * 8;
  int col = n0 + n;
  float bi = bsum[0 * 256 + col], bf = bsum[1 * 256 + col];
  float bg = bsum[2 * 256 + col], bo = bsum[3 * 256 + col];
#pragma unroll
  for (int r = 0; r < 8; ++r) {
    int row = m0 + mb + r;
    float iv = acc[0][r] + bi;
    float fv = acc[1][r] + bf;
    float gv = acc[2][r] + bg;
    float ov = acc[3][r] + bo;
    float c = sigm(fv) * cprev[row * 256 + col] + sigm(iv) * tanhf(gv);
    float h = sigm(ov) * tanhf(c);
    cout[row * 256 + col] = c;
    hout_bf[(size_t)row * ldhob + col] = f2bf_u(h);
    if (hout_f) hout_f[(size_t)row * ldhof + col] = h;
  }
}

// ---------------------------------------------------------------------------
// Small helper kernels
// ---------------------------------------------------------------------------
__global__ void cvt_f32_bf16(const float* __restrict__ in, unsigned short* __restrict__ out, int n) {
  int i = blockIdx.x * blockDim.x + threadIdx.x;
  if (i < n) out[i] = f2bf_u(in[i]);
}
__global__ void bias_sum_kernel(const float* a, const float* b, float* out, int n) {
  int i = blockIdx.x * blockDim.x + threadIdx.x;
  if (i < n) out[i] = a[i] + b[i];
}
__global__ void zero_us(unsigned short* p, int n) {
  int i = blockIdx.x * blockDim.x + threadIdx.x;
  if (i < n) p[i] = 0;
}
__global__ void zero_f(float* p, int n) {
  int i = blockIdx.x * blockDim.x + threadIdx.x;
  if (i < n) p[i] = 0.0f;
}

// enc_out = 0.5*(y1f + y1b) (time-major inputs [S,B,H] -> [B,S,H]) and
// x2[b,s] = enc_out[b,s,:] . a2W
__global__ void enc_combine_kernel(const float* __restrict__ y1f, const float* __restrict__ y1b,
                                   const float* __restrict__ a2W,
                                   float* __restrict__ enc_out, float* __restrict__ x2) {
  int bs = blockIdx.x;                // = b*20 + s
  int b = bs / 20, s = bs % 20, t = threadIdx.x;
  __shared__ float red[256];
  float v = 0.5f * (y1f[(s * 256 + b) * 256 + t] + y1b[(s * 256 + b) * 256 + t]);
  enc_out[(size_t)bs * 256 + t] = v;
  red[t] = v * a2W[t];
  __syncthreads();
  for (int r = 128; r > 0; r >>= 1) { if (t < r) red[t] += red[t + r]; __syncthreads(); }
  if (t == 0) x2[bs] = red[0];
}

// Decoder initial states from encoder finals
__global__ void init_states_kernel(const unsigned short* __restrict__ l0cat,
                                   const float* __restrict__ c0f, const float* __restrict__ c0b,
                                   const float* __restrict__ h1f_last, const float* __restrict__ h1b_first,
                                   const float* __restrict__ c1f, const float* __restrict__ c1b,
                                   unsigned short* h1_bf, float* c1o,
                                   unsigned short* h2_bf, float* c2o) {
  int b = blockIdx.x, t = threadIdx.x;
  int idx = b * 256 + t;
  float h0f = bfu2f(l0cat[((size_t)19 * 256 + b) * 512 + t]);
  float h0b = bfu2f(l0cat[((size_t)0 * 256 + b) * 512 + 256 + t]);
  h1_bf[idx] = f2bf_u(0.5f * (h0f + h0b));
  c1o[idx] = 0.5f * (c0f[idx] + c0b[idx]);
  float hf = h1f_last[idx], hb = h1b_first[idx];
  h2_bf[idx] = f2bf_u(0.5f * (hf + hb));
  c2o[idx] = 0.5f * (c1f[idx] + c1b[idx]);
}

// dec_init rows + dec_next tail of output
__global__ void build_dec_init_kernel(const float* __restrict__ first,
                                      const float* __restrict__ dec_input,
                                      float* __restrict__ dec_init, float* __restrict__ dnext) {
  int i = blockIdx.x * blockDim.x + threadIdx.x;
  if (i < 256 * 5) {
    int b = i / 5, d = i % 5;
    dec_init[i] = (b == 0) ? first[d] : dec_input[((b - 1) * 20 + 19) * 5 + d];
  } else if (i < 256 * 5 + 5) {
    int d = i - 256 * 5;
    dnext[d] = dec_input[(255 * 20 + 19) * 5 + d];
  }
}

// e = dec_in @ d_emb_W^T + d_emb_b  (K=5) -> bf16
__global__ void demb_kernel(const float* __restrict__ din, int ld,
                            const float* __restrict__ W, const float* __restrict__ bias,
                            unsigned short* __restrict__ e) {
  int b = blockIdx.x, h = threadIdx.x;
  float s = bias[h];
#pragma unroll
  for (int d = 0; d < 5; ++d) s += din[b * ld + d] * W[h * 5 + d];
  e[b * 256 + h] = f2bf_u(s);
}

// Attention + output projection for one decoder step j. One block per batch row.
__global__ void attn_out_kernel(const float* __restrict__ h2f, const float* __restrict__ a1W,
                                const float* __restrict__ x2, const float* __restrict__ enc_out,
                                const float* __restrict__ rc_all, int j,
                                const float* __restrict__ linW, const float* __restrict__ linb,
                                float* __restrict__ out) {
  int b = blockIdx.x, t = threadIdx.x;
  __shared__ float red[256];
  __shared__ float w_s[20];
  __shared__ float ctx_s[256];
  float hv = h2f[b * 256 + t];
  red[t] = hv * a1W[t];
  __syncthreads();
  for (int r = 128; r > 0; r >>= 1) { if (t < r) red[t] += red[t + r]; __syncthreads(); }
  float x1 = red[0];
  if (t == 0) {
    float sc[20]; float mx = -1e30f;
    for (int s = 0; s < 20; ++s) { sc[s] = tanhf(x1 + x2[b * 20 + s]); mx = fmaxf(mx, sc[s]); }
    float sum = 0.f;
    for (int s = 0; s < 20; ++s) { sc[s] = __expf(sc[s] - mx); sum += sc[s]; }
    float inv = 1.0f / sum;
    for (int s = 0; s < 20; ++s) w_s[s] = sc[s] * inv;
  }
  __syncthreads();
  float ctx = 0.f;
  for (int s = 0; s < 20; ++s) ctx += w_s[s] * enc_out[((size_t)b * 20 + s) * 256 + t];
  ctx_s[t] = ctx;
  __syncthreads();
  float acc[5] = {0.f, 0.f, 0.f, 0.f, 0.f};
#pragma unroll
  for (int q = 0; q < 4; ++q) {
    int k = q * 256 + t;
    float cv;
    if (k < 256)      cv = h2f[b * 256 + k];
    else if (k < 512) cv = ctx_s[k - 256];
    else              cv = rc_all[((size_t)b * 20 + j) * 512 + (k - 512)];
#pragma unroll
    for (int d = 0; d < 5; ++d) acc[d] += cv * linW[d * 1024 + k];
  }
  for (int d = 0; d < 5; ++d) {
    __syncthreads();
    red[t] = acc[d];
    __syncthreads();
    for (int r = 128; r > 0; r >>= 1) { if (t < r) red[t] += red[t + r]; __syncthreads(); }
    if (t == 0) out[((size_t)b * 20 + j) * 5 + d] = red[0] + linb[d];
  }
}

// ---------------------------------------------------------------------------
// Host orchestration
// ---------------------------------------------------------------------------
extern "C" void kernel_launch(void* const* d_in, const int* in_sizes, int n_in,
                              void* d_out, int out_size, void* d_ws, size_t ws_size,
                              hipStream_t stream) {
  (void)in_sizes; (void)n_in; (void)out_size; (void)ws_size;
  const int BSZ = 256, S = 20, H = 256, DIN = 4032;
  const int MROWS = BSZ * S;   // 5120

  const float* d_first  = (const float*)d_in[0];
  const float* d_decin  = (const float*)d_in[1];
  const float* d_cnn    = (const float*)d_in[2];
  const float* d_ebW    = (const float*)d_in[3];
  const float* d_ebb    = (const float*)d_in[4];
  // LSTM param sets: 0=enc l0f, 1=enc l0b, 2=enc l1f, 3=enc l1b, 4=dec l0, 5=dec l1
  const float* Wih[6]; const float* Whh[6]; const float* bih[6]; const float* bhh[6];
  for (int i = 0; i < 6; ++i) {
    Wih[i] = (const float*)d_in[5 + i * 4 + 0];
    Whh[i] = (const float*)d_in[5 + i * 4 + 1];
    bih[i] = (const float*)d_in[5 + i * 4 + 2];
    bhh[i] = (const float*)d_in[5 + i * 4 + 3];
  }
  const float* d_a1W  = (const float*)d_in[29];
  const float* d_a2W  = (const float*)d_in[30];
  const float* d_dW   = (const float*)d_in[31];
  const float* d_dB   = (const float*)d_in[32];
  const float* d_rcW  = (const float*)d_in[33];
  const float* d_rcb  = (const float*)d_in[34];
  const float* d_linW = (const float*)d_in[35];
  const float* d_linb = (const float*)d_in[36];

  float* out = (float*)d_out;

  // ---- workspace layout
  char* ws = (char*)d_ws;
  size_t off = 0;
  auto alloc = [&](size_t bytes) -> char* {
    char* p = ws + off;
    off += (bytes + 255) & ~(size_t)255;
    return p;
  };
  unsigned short* cnn_bf = (unsigned short*)alloc((size_t)MROWS * DIN * 2);
  unsigned short* ebW_bf = (unsigned short*)alloc((size_t)H * DIN * 2);
  unsigned short* rcW_bf = (unsigned short*)alloc((size_t)2 * H * DIN * 2);
  const int DxArr[6] = {256, 256, 512, 512, 256, 256};
  unsigned short* Wih_bf[6]; unsigned short* Whh_bf[6]; float* bsum[6];
  for (int i = 0; i < 6; ++i) {
    Wih_bf[i] = (unsigned short*)alloc((size_t)1024 * DxArr[i] * 2);
    Whh_bf[i] = (unsigned short*)alloc((size_t)1024 * 256 * 2);
    bsum[i]   = (float*)alloc(1024 * 4);
  }
  unsigned short* emb_bf = (unsigned short*)alloc((size_t)MROWS * H * 2);
  float* rc_all          = (float*)alloc((size_t)MROWS * 512 * 4);
  unsigned short* l0cat  = (unsigned short*)alloc((size_t)S * BSZ * 512 * 2);
  float* y1f_seq         = (float*)alloc((size_t)S * BSZ * H * 4);
  float* y1b_seq         = (float*)alloc((size_t)S * BSZ * H * 4);
  float* enc_out         = (float*)alloc((size_t)MROWS * H * 4);
  float* x2              = (float*)alloc((size_t)MROWS * 4);
  unsigned short* zero_h = (unsigned short*)alloc((size_t)BSZ * H * 2);
  float* czero           = (float*)alloc((size_t)BSZ * H * 4);
  float* cpp[4][2];
  for (int i = 0; i < 4; ++i) for (int p = 0; p < 2; ++p)
    cpp[i][p] = (float*)alloc((size_t)BSZ * H * 4);
  unsigned short* h_l1f_pp[2]; unsigned short* h_l1b_pp[2];
  for (int p = 0; p < 2; ++p) h_l1f_pp[p] = (unsigned short*)alloc((size_t)BSZ * H * 2);
  for (int p = 0; p < 2; ++p) h_l1b_pp[p] = (unsigned short*)alloc((size_t)BSZ * H * 2);
  unsigned short* init_h1 = (unsigned short*)alloc((size_t)BSZ * H * 2);
  float* init_c1          = (float*)alloc((size_t)BSZ * H * 4);
  unsigned short* init_h2 = (unsigned short*)alloc((size_t)BSZ * H * 2);
  float* init_c2          = (float*)alloc((size_t)BSZ * H * 4);
  unsigned short* h1pp[2]; float* c1pp[2]; unsigned short* h2pp[2]; float* c2pp[2];
  for (int p = 0; p < 2; ++p) { h1pp[p] = (unsigned short*)alloc((size_t)BSZ * H * 2); }
  for (int p = 0; p < 2; ++p) { c1pp[p] = (float*)alloc((size_t)BSZ * H * 4); }
  for (int p = 0; p < 2; ++p) { h2pp[p] = (unsigned short*)alloc((size_t)BSZ * H * 2); }
  for (int p = 0; p < 2; ++p) { c2pp[p] = (float*)alloc((size_t)BSZ * H * 4); }
  float* h2_f             = (float*)alloc((size_t)BSZ * H * 4);
  unsigned short* e_bf    = (unsigned short*)alloc((size_t)BSZ * H * 2);
  float* dec_init         = (float*)alloc((size_t)BSZ * 5 * 4);

  auto launch_cvt = [&](const float* src, unsigned short* dst, int n) {
    cvt_f32_bf16<<<(n + 255) / 256, 256, 0, stream>>>(src, dst, n);
  };

  // ---- 1. precision conversion of weights + activations
  launch_cvt(d_cnn, cnn_bf, MROWS * DIN);
  launch_cvt(d_ebW, ebW_bf, H * DIN);
  launch_cvt(d_rcW, rcW_bf, 2 * H * DIN);
  for (int i = 0; i < 6; ++i) {
    launch_cvt(Wih[i], Wih_bf[i], 1024 * DxArr[i]);
    launch_cvt(Whh[i], Whh_bf[i], 1024 * 256);
    bias_sum_kernel<<<4, 256, 0, stream>>>(bih[i], bhh[i], bsum[i], 1024);
  }
  zero_us<<<(BSZ * H + 255) / 256, 256, 0, stream>>>(zero_h, BSZ * H);
  zero_f<<<(BSZ * H + 255) / 256, 256, 0, stream>>>(czero, BSZ * H);

  // ---- 2. big WMMA GEMMs (hoisted out of the recurrence)
  {
    dim3 g(H / 16, MROWS / 64);
    gemm_bf16_kernel<<<g, 128, 0, stream>>>(cnn_bf, DIN, ebW_bf, DIN, d_ebb,
                                            emb_bf, H, (float*)nullptr, 0, DIN);
  }
  {
    dim3 g(2 * H / 16, MROWS / 64);
    gemm_bf16_kernel<<<g, 128, 0, stream>>>(cnn_bf, DIN, rcW_bf, DIN, d_rcb,
                                            (unsigned short*)nullptr, 0, rc_all, 512, DIN);
  }

  dim3 gl(16, 4); // LSTM tile grid: 16 n-tiles x 4 blocks x 4 waves = 256 tiles

  // ---- 3. encoder layer 0 forward (writes l0cat[:, :, 0:256])
  for (int s = 0; s < S; ++s) {
    const unsigned short* hp = s ? (l0cat + (size_t)(s - 1) * BSZ * 512) : zero_h;
    int ldh = s ? 512 : 256;
    lstm_cell_kernel<<<gl, 128, 0, stream>>>(
        emb_bf + s * H, S * H, 256, Wih_bf[0], hp, ldh, Whh_bf[0], bsum[0],
        s ? cpp[0][(s - 1) & 1] : czero, cpp[0][s & 1],
        l0cat + (size_t)s * BSZ * 512, 512, (float*)nullptr, 0);
  }
  // ---- encoder layer 0 backward (writes l0cat[:, :, 256:512])
  for (int t = 0; t < S; ++t) {
    int s = S - 1 - t;
    const unsigned short* hp = t ? (l0cat + (size_t)(s + 1) * BSZ * 512 + 256) : zero_h;
    int ldh = t ? 512 : 256;
    lstm_cell_kernel<<<gl, 128, 0, stream>>>(
        emb_bf + s * H, S * H, 256, Wih_bf[1], hp, ldh, Whh_bf[1], bsum[1],
        t ? cpp[1][(t - 1) & 1] : czero, cpp[1][t & 1],
        l0cat + (size_t)s * BSZ * 512 + 256, 512, (float*)nullptr, 0);
  }
  // ---- encoder layer 1 forward (Dx=512)
  for (int s = 0; s < S; ++s) {
    const unsigned short* hp = s ? h_l1f_pp[(s - 1) & 1] : zero_h;
    lstm_cell_kernel<<<gl, 128, 0, stream>>>(
        l0cat + (size_t)s * BSZ * 512, 512, 512, Wih_bf[2], hp, 256, Whh_bf[2], bsum[2],
        s ? cpp[2][(s - 1) & 1] : czero, cpp[2][s & 1],
        h_l1f_pp[s & 1], 256, y1f_seq + (size_t)s * BSZ * H, 256);
  }
  // ---- encoder layer 1 backward
  for (int t = 0; t < S; ++t) {
    int s = S - 1 - t;
    const unsigned short* hp = t ? h_l1b_pp[(t - 1) & 1] : zero_h;
    lstm_cell_kernel<<<gl, 128, 0, stream>>>(
        l0cat + (size_t)s * BSZ * 512, 512, 512, Wih_bf[3], hp, 256, Whh_bf[3], bsum[3],
        t ? cpp[3][(t - 1) & 1] : czero, cpp[3][t & 1],
        h_l1b_pp[t & 1], 256, y1b_seq + (size_t)s * BSZ * H, 256);
  }

  // ---- 4. enc_out, x2, decoder initial states, dec_init/dec_next
  enc_combine_kernel<<<MROWS, 256, 0, stream>>>(y1f_seq, y1b_seq, d_a2W, enc_out, x2);
  init_states_kernel<<<BSZ, 256, 0, stream>>>(
      l0cat, cpp[0][(S - 1) & 1], cpp[1][(S - 1) & 1],
      y1f_seq + (size_t)(S - 1) * BSZ * H, y1b_seq,
      cpp[2][(S - 1) & 1], cpp[3][(S - 1) & 1],
      init_h1, init_c1, init_h2, init_c2);
  build_dec_init_kernel<<<(256 * 5 + 5 + 255) / 256, 256, 0, stream>>>(
      d_first, d_decin, dec_init, out + BSZ * S * 5);

  // ---- 5. decoder loop
  for (int j = 0; j < S; ++j) {
    const float* din_src = (j == 0) ? dec_init : (out + (j - 1) * 5);
    int din_ld = (j == 0) ? 5 : S * 5;
    demb_kernel<<<BSZ, 256, 0, stream>>>(din_src, din_ld, d_dW, d_dB, e_bf);

    const unsigned short* h1r = (j == 0) ? init_h1 : h1pp[(j - 1) & 1];
    const float* c1r = (j == 0) ? init_c1 : c1pp[(j - 1) & 1];
    lstm_cell_kernel<<<gl, 128, 0, stream>>>(
        e_bf, 256, 256, Wih_bf[4], h1r, 256, Whh_bf[4], bsum[4],
        c1r, c1pp[j & 1], h1pp[j & 1], 256, (float*)nullptr, 0);

    const unsigned short* h2r = (j == 0) ? init_h2 : h2pp[(j - 1) & 1];
    const float* c2r = (j == 0) ? init_c2 : c2pp[(j - 1) & 1];
    lstm_cell_kernel<<<gl, 128, 0, stream>>>(
        h1pp[j & 1], 256, 256, Wih_bf[5], h2r, 256, Whh_bf[5], bsum[5],
        c2r, c2pp[j & 1], h2pp[j & 1], 256, h2_f, 256);

    attn_out_kernel<<<BSZ, 256, 0, stream>>>(h2_f, d_a1W, x2, enc_out, rc_all, j,
                                             d_linW, d_linb, out);
  }
}
